// ConvLRULayer_40492951667441
// MI455X (gfx1250) — compile-verified, matrix operand order checked by reading, not run on hardware
//
#include <hip/hip_runtime.h>
#include <hip/hip_bf16.h>

// ---------------------------------------------------------------------------
// ConvLRULayer for MI455X (gfx1250, wave32, WMMA 16x16x32 bf16)
// B=4, L=32, C=32, S=64, W=64, R=32, MH=32
// - All WMMA operands staged K-fastest => fragments load as 2x ds_load_b128.
// - Conversion-free LDS staging uses GLOBAL_LOAD_ASYNC_TO_LDS_B128 (ASYNCcnt).
// ---------------------------------------------------------------------------

#define Bb 4
#define Ll 32
#define Cc 32
#define Ss 64
#define Ww 64
#define Rr 32
#define MHh 32

typedef __attribute__((ext_vector_type(16))) __bf16 v16bf;
typedef __attribute__((ext_vector_type(8)))  float  v8f;

static __device__ inline v8f wmma_bf16(v16bf a, v16bf b, v8f c) {
  // v_wmma_f32_16x16x32_bf16: (neg_a, A, neg_b, B, c_mod, C, reuse_a, reuse_b)
  return __builtin_amdgcn_wmma_f32_16x16x32_bf16(false, a, false, b, (short)0, c,
                                                 false, false);
}

// 16-bit A/B fragment, K contiguous in memory at p (p -> element K = kh*8):
// elements 0..7 = K kh*8+0..7 ; elements 8..15 = K 16+kh*8+0..7  (ISA 7.12.2)
static __device__ inline v16bf frag_contig(const __bf16* p) {
  v16bf v;
#pragma unroll
  for (int j = 0; j < 8; ++j) v[j] = p[j];
#pragma unroll
  for (int j = 0; j < 8; ++j) v[8 + j] = p[16 + j];
  return v;
}

static __device__ inline v8f v8f_zero() {
  v8f z = {0.f, 0.f, 0.f, 0.f, 0.f, 0.f, 0.f, 0.f};
  return z;
}

// Async 16-byte global->LDS copy (no VGPR data path, tracked by ASYNCcnt).
static __device__ inline void async_copy_b128(void* lds_dst, const void* gsrc) {
  unsigned laddr = (unsigned)(size_t)lds_dst;          // LDS byte address
  unsigned long long gaddr = (unsigned long long)(size_t)gsrc;
  asm volatile("global_load_async_to_lds_b128 %0, %1, off"
               :
               : "v"(laddr), "v"(gaddr)
               : "memory");
}

static __device__ inline void wait_async0() {
  asm volatile("s_wait_asynccnt 0" ::: "memory");
}

// ---------------------------------------------------------------------------
// Kernel 1: fold fuse∘(convr,convi)∘proj into one 3x3 conv with 64 in-chans.
//   Wcat[o][tap][k], k<32 -> weight on Re(y)[k], k>=32 -> weight on Im(y)[k-32]
//   B0[o] = fuse_b[o] + sum_i fw[o,i]*convr_b[i] + fw[o,C+i]*convi_b[i]
// ---------------------------------------------------------------------------
__global__ __launch_bounds__(256) void prep_kernel(
    const float* __restrict__ pWre, const float* __restrict__ pWim,
    const float* __restrict__ convr_w, const float* __restrict__ convr_b,
    const float* __restrict__ convi_w, const float* __restrict__ convi_b,
    const float* __restrict__ fuse_w, const float* __restrict__ fuse_b,
    float* __restrict__ effr, float* __restrict__ effi,
    __bf16* __restrict__ wcat, float* __restrict__ b0) {
  int t = threadIdx.x;
  for (int idx = t; idx < 32 * 32 * 9; idx += 256) {
    int tap = idx % 9;
    int cch = (idx / 9) & 31;
    int o = idx / (9 * 32);
    float sr = 0.f, si = 0.f;
    for (int i = 0; i < 32; ++i) {
      sr += fuse_w[o * 64 + i] * convr_w[(i * 32 + cch) * 9 + tap];
      si += fuse_w[o * 64 + 32 + i] * convi_w[(i * 32 + cch) * 9 + tap];
    }
    effr[idx] = sr;
    effi[idx] = si;
  }
  __syncthreads();
  for (int idx = t; idx < 32 * 32 * 9; idx += 256) {
    int tap = idx % 9;
    int cp = (idx / 9) & 31;
    int o = idx / (9 * 32);
    float wr = 0.f, wi = 0.f;
    for (int c = 0; c < 32; ++c) {
      float er = effr[(o * 32 + c) * 9 + tap];
      float ei = effi[(o * 32 + c) * 9 + tap];
      float pr = pWre[c * 32 + cp];
      float pi = pWim[c * 32 + cp];
      wr += er * pr + ei * pi;
      wi += -er * pi + ei * pr;
    }
    wcat[o * 576 + tap * 64 + cp] = (__bf16)wr;
    wcat[o * 576 + tap * 64 + 32 + cp] = (__bf16)wi;
  }
  if (t < 32) {
    float s = fuse_b[t];
    for (int i = 0; i < 32; ++i)
      s += fuse_w[t * 64 + i] * convr_b[i] + fuse_w[t * 64 + 32 + i] * convi_b[i];
    b0[t] = s;
  }
}

// ---------------------------------------------------------------------------
// Kernel 2: per (b,l): ctx mean over (S,W), MLP, lambda poles -> lam_{re,im}
// ---------------------------------------------------------------------------
__global__ __launch_bounds__(256) void mlp_lam_kernel(
    const float* __restrict__ x, const float* __restrict__ dt,
    const float* __restrict__ plb, const float* __restrict__ disp,
    const float* __restrict__ w1, const float* __restrict__ b1,
    const float* __restrict__ w2, const float* __restrict__ b2,
    const float* __restrict__ fscale, float* __restrict__ lamre,
    float* __restrict__ lamim) {
  __shared__ float partial[32][8];
  __shared__ float inp[33];
  __shared__ float hid[32];
  int bl = blockIdx.x;
  int t = threadIdx.x;
  int c = t >> 3, sl = t & 7;
  const float* xp = x + (size_t)bl * Cc * Ss * Ww + (size_t)c * Ss * Ww;
  float s = 0.f;
  for (int i = sl; i < Ss * Ww; i += 8) s += xp[i];
  partial[c][sl] = s;
  __syncthreads();
  if (t < 32) {
    float sum = 0.f;
    for (int j = 0; j < 8; ++j) sum += partial[t][j];
    inp[t] = sum * (1.f / (Ss * Ww));
  }
  if (t == 0) inp[32] = dt[bl];
  __syncthreads();
  if (t < 32) {
    float h = b1[t];
    for (int cc = 0; cc < 33; ++cc) h += inp[cc] * w1[cc * 32 + t];
    hid[t] = tanhf(h);
  }
  __syncthreads();
  float fs = *fscale;
  for (int pi = t; pi < Cc * Rr; pi += 256) {
    int j0 = pi * 2;
    float m0 = b2[j0], m1 = b2[j0 + 1];
    for (int mh = 0; mh < 32; ++mh) {
      float h = hid[mh];
      m0 += h * w2[mh * 2048 + j0];
      m1 += h * w2[mh * 2048 + j0 + 1];
    }
    float dnu = fs * tanhf(m0);
    float dth = fs * tanhf(m1);
    float nu = plb[pi] + disp[pi];
    float th = plb[1024 + pi] + disp[1024 + pi];
    float mag = expf(-expf(nu + dnu));
    float ang = expf(th + dth);
    lamre[(size_t)bl * 1024 + pi] = mag * cosf(ang);
    lamim[(size_t)bl * 1024 + pi] = mag * sinf(ang);
  }
}

// ---------------------------------------------------------------------------
// Kernel 3: encode. Per (b,l,c):  P = Ure^T X, Q = Uim^T X  (WMMA bf16, R x W)
//   z_re[r] = sum_w P[r,w]Vre[w,r] - Q[r,w]Vim[w,r]
//   z_im[r] = -sum_w P[r,w]Vim[w,r] + Q[r,w]Vre[w,r]
// X/U staged transposed (K = s fastest); V staged via async b128 copies.
// ---------------------------------------------------------------------------
__global__ __launch_bounds__(256) void encode_kernel(
    const float* __restrict__ x, const float* __restrict__ Ure,
    const float* __restrict__ Uim, const float* __restrict__ Vre,
    const float* __restrict__ Vim, float* __restrict__ zre,
    float* __restrict__ zim) {
  __shared__ __align__(16) __bf16 sXt[Ww * Ss];   // [w][s]
  __shared__ __align__(16) __bf16 sUrT[Rr * Ss];  // [r][s]
  __shared__ __align__(16) __bf16 sUiT[Rr * Ss];  // [r][s]
  __shared__ __align__(16) float sVr[Ww * Rr];    // [w][r]
  __shared__ __align__(16) float sVi[Ww * Rr];
  __shared__ float szr[Rr], szi[Rr];
  int blc = blockIdx.x;  // (b*L+l)*C + c
  int c = blc & (Cc - 1);
  int t = threadIdx.x;
  const float* xp = x + (size_t)blc * (Ss * Ww);
  __builtin_prefetch(xp, 0, 1);  // global_prefetch_b8 on streaming tile
  // V tiles are conversion-free copies -> async DMA to LDS (ASYNCcnt)
  const float* vpr = Vre + (size_t)c * (Ww * Rr);
  const float* vpi = Vim + (size_t)c * (Ww * Rr);
  for (int ch = t; ch < (Ww * Rr) / 4; ch += 256) {  // 512 x 16B per array
    async_copy_b128(&sVr[ch * 4], vpr + ch * 4);
    async_copy_b128(&sVi[ch * 4], vpi + ch * 4);
  }
  for (int i = t; i < Ss * Ww; i += 256) {
    int s = i >> 6, w = i & 63;
    sXt[w * Ss + s] = (__bf16)xp[i];
  }
  const float* upr = Ure + (size_t)c * (Ss * Rr);
  const float* upi = Uim + (size_t)c * (Ss * Rr);
  for (int i = t; i < Ss * Rr; i += 256) {
    int s = i >> 5, r = i & 31;
    sUrT[r * Ss + s] = (__bf16)upr[i];
    sUiT[r * Ss + s] = (__bf16)upi[i];
  }
  if (t < Rr) {
    szr[t] = 0.f;
    szi[t] = 0.f;
  }
  wait_async0();
  __syncthreads();

  int wid = t >> 5, lane = t & 31, n15 = lane & 15, kh = lane >> 4;
  int mtile = wid & 1, ntile = wid >> 1;  // 2 x 4 output tiles of 16x16
  v8f p = v8f_zero(), q = v8f_zero();
#pragma unroll
  for (int ks = 0; ks < 2; ++ks) {
    int koff = ks * 32;
    v16bf bx = frag_contig(sXt + (ntile * 16 + n15) * Ss + koff + kh * 8);
    v16bf ar = frag_contig(sUrT + (mtile * 16 + n15) * Ss + koff + kh * 8);
    v16bf ai = frag_contig(sUiT + (mtile * 16 + n15) * Ss + koff + kh * 8);
    p = wmma_bf16(ar, bx, p);
    q = wmma_bf16(ai, bx, q);
  }
  int wcol = ntile * 16 + n15;
  float crv[8], civ[8];
#pragma unroll
  for (int i = 0; i < 8; ++i) {
    int r = mtile * 16 + kh * 8 + i;
    float vr = sVr[wcol * Rr + r];
    float vi = sVi[wcol * Rr + r];
    crv[i] = p[i] * vr - q[i] * vi;
    civ[i] = -(p[i] * vi + q[i] * vr);
  }
#pragma unroll
  for (int off = 1; off < 16; off <<= 1) {
#pragma unroll
    for (int i = 0; i < 8; ++i) {
      crv[i] += __shfl_xor(crv[i], off, 32);
      civ[i] += __shfl_xor(civ[i], off, 32);
    }
  }
  if (n15 == 0) {
#pragma unroll
    for (int i = 0; i < 8; ++i) {
      int r = mtile * 16 + kh * 8 + i;
      atomicAdd(&szr[r], crv[i]);
      atomicAdd(&szi[r], civ[i]);
    }
  }
  __syncthreads();
  if (t < Rr) {
    zre[(size_t)blc * Rr + t] = szr[t];
    zim[(size_t)blc * Rr + t] = szi[t];
  }
}

// ---------------------------------------------------------------------------
// Kernel 4: diagonal complex recurrence over L (tiny, sequential in L)
// ---------------------------------------------------------------------------
__global__ __launch_bounds__(256) void scan_kernel(
    const float* __restrict__ lamre, const float* __restrict__ lamim,
    const float* __restrict__ zre, const float* __restrict__ zim,
    float* __restrict__ hre, float* __restrict__ him) {
  int id = blockIdx.x * 256 + threadIdx.x;  // b*1024 + c*32 + r
  int b = id >> 10;
  int cr = id & 1023;
  float hr = 0.f, hi = 0.f;
  for (int l = 0; l < Ll; ++l) {
    size_t idx = ((size_t)(b * Ll + l)) * 1024 + cr;
    float lr = lamre[idx], li = lamim[idx];
    float nr = lr * hr - li * hi + zre[idx];
    float ni = lr * hi + li * hr + zim[idx];
    hr = nr;
    hi = ni;
    hre[idx] = hr;
    him[idx] = hi;
  }
}

// ---------------------------------------------------------------------------
// Kernel 5: decode. Per (b,l,c):  Y = (U diag(h)) V^T  (complex, 4 real GEMMs)
//   Output goes to channel-interleaved field ycat[bl][s][w][k64]
//   (k<32 = Re channel k (+pb_re), k>=32 = Im channel k-32 (+pb_im)),
//   which is the exact im2col K-fastest layout the conv kernel wants.
//   bf16 WMMA has no A/B NEG -> stage -Vim to realize Yre = AreBre + Aim(-Bim)
// ---------------------------------------------------------------------------
__global__ __launch_bounds__(256) void decode_kernel(
    const float* __restrict__ Ure, const float* __restrict__ Uim,
    const float* __restrict__ Vre, const float* __restrict__ Vim,
    const float* __restrict__ hre, const float* __restrict__ him,
    const float* __restrict__ pbre, const float* __restrict__ pbim,
    __bf16* __restrict__ ycat) {
  __shared__ __align__(16) __bf16 sAr[Ss * Rr];   // [s][r]  (K=r fastest)
  __shared__ __align__(16) __bf16 sAi[Ss * Rr];
  __shared__ __align__(16) __bf16 sBr[Ww * Rr];   // [w][r]  (K=r fastest)
  __shared__ __align__(16) __bf16 sBi[Ww * Rr];
  __shared__ __align__(16) __bf16 sBin[Ww * Rr];  // -Vim
  __shared__ float shr[Rr], shi[Rr];
  int blc = blockIdx.x;
  int c = blc & (Cc - 1);
  int bl = blc >> 5;
  int t = threadIdx.x;
  if (t < Rr) {
    shr[t] = hre[(size_t)blc * Rr + t];
    shi[t] = him[(size_t)blc * Rr + t];
  }
  __syncthreads();
  for (int i = t; i < Ss * Rr; i += 256) {
    int r = i & 31;
    float ur = Ure[(size_t)c * Ss * Rr + i];
    float ui = Uim[(size_t)c * Ss * Rr + i];
    sAr[i] = (__bf16)(ur * shr[r] - ui * shi[r]);
    sAi[i] = (__bf16)(ur * shi[r] + ui * shr[r]);
  }
  // V is stored [w][r] in global: already the transposed-B layout -> copy
  for (int i = t; i < Ww * Rr; i += 256) {
    float vr = Vre[(size_t)c * Ww * Rr + i];
    float vi = Vim[(size_t)c * Ww * Rr + i];
    sBr[i] = (__bf16)vr;
    sBi[i] = (__bf16)vi;
    sBin[i] = (__bf16)(-vi);
  }
  __syncthreads();
  int wid = t >> 5, lane = t & 31, n15 = lane & 15, kh = lane >> 4;
  float pbr = pbre[c], pbi = pbim[c];
#pragma unroll
  for (int sub = 0; sub < 2; ++sub) {
    int tile = wid * 2 + sub;  // 0..15 of the 4x4 tile grid
    int mt = tile >> 2, nt = tile & 3;
    v16bf ar = frag_contig(sAr + (mt * 16 + n15) * Rr + kh * 8);
    v16bf ai = frag_contig(sAi + (mt * 16 + n15) * Rr + kh * 8);
    v16bf br = frag_contig(sBr + (nt * 16 + n15) * Rr + kh * 8);
    v16bf bi = frag_contig(sBi + (nt * 16 + n15) * Rr + kh * 8);
    v16bf bin = frag_contig(sBin + (nt * 16 + n15) * Rr + kh * 8);
    v8f accr = wmma_bf16(ar, br, v8f_zero());
    accr = wmma_bf16(ai, bin, accr);
    v8f acci = wmma_bf16(ar, bi, v8f_zero());
    acci = wmma_bf16(ai, br, acci);
    int w = nt * 16 + n15;
#pragma unroll
    for (int i = 0; i < 8; ++i) {
      int sidx = mt * 16 + kh * 8 + i;
      size_t base = (((size_t)bl * Ss + sidx) * Ww + w) * 64;
      ycat[base + c] = (__bf16)(accr[i] + pbr);
      ycat[base + 32 + c] = (__bf16)(acci[i] + pbi);
    }
  }
}

// ---------------------------------------------------------------------------
// Kernel 6: fused 3x3 conv over (S,W) via im2col WMMA.
//   out[o,s,w] = B0[o] + sum_{tap,k<64} Wcat[o][tap][k] * ycat[s+ky-1, w+kx-1, k]
//   Block: one (b,l) x 4 S-rows. LDS tile pre-padded in w (66 cols), 6 rows.
//   Interior cells staged via async b128 DMA (128B per cell = 8 transfers);
//   border cells zero-filled with plain LDS stores.
// ---------------------------------------------------------------------------
__global__ __launch_bounds__(256) void conv_kernel(
    const __bf16* __restrict__ ycat, const __bf16* __restrict__ wcat,
    const float* __restrict__ b0, float* __restrict__ convout) {
  __shared__ __align__(16) __bf16 sY[6 * 66 * 64];  // [row6][col66][k64] 50688B
  int blk = blockIdx.x;
  int chunk = blk & 15;  // 16 chunks of 4 S-rows
  int bl = blk >> 4;     // b*L + l
  int s0 = chunk * 4;
  int t = threadIdx.x;
  for (int cell = t; cell < 6 * 66; cell += 256) {
    int colp = cell % 66;
    int row = cell / 66;
    int col = colp - 1;
    int sg = s0 - 1 + row;
    __bf16* dst = &sY[cell * 64];
    if (col >= 0 && col < Ww && sg >= 0 && sg < Ss) {
      const __bf16* src = ycat + (((size_t)bl * Ss + sg) * Ww + col) * 64;
#pragma unroll
      for (int qv = 0; qv < 8; ++qv)
        async_copy_b128(dst + qv * 8, src + qv * 8);  // 8 x 16B = 128B
    } else {
#pragma unroll
      for (int qv = 0; qv < 32; ++qv) ((int*)dst)[qv] = 0;
    }
  }
  wait_async0();
  __syncthreads();
  int wid = t >> 5, lane = t & 31, n15 = lane & 15, kh = lane >> 4;
  int mt = wid & 1;
  int nbase = wid >> 1;  // 0..3
  v8f acc[4] = {v8f_zero(), v8f_zero(), v8f_zero(), v8f_zero()};
  for (int tap = 0; tap < 9; ++tap) {
    int ky = tap / 3, kx = tap % 3;
#pragma unroll
    for (int ks = 0; ks < 2; ++ks) {
      // A fragment from global Wcat (tiny, L2-resident): contiguous in k
      const __bf16* ap =
          wcat + (mt * 16 + n15) * 576 + tap * 64 + ks * 32 + kh * 8;
      v16bf a = frag_contig(ap);
#pragma unroll
      for (int sub = 0; sub < 4; ++sub) {
        int ntile = nbase + 4 * sub;  // 0..15
        int P = ntile * 16 + n15;     // spatial position in chunk
        int srl = P >> 6;             // local row 0..3
        int wpos = P & 63;
        int rowIdx = srl + ky;        // staged row 0..5
        int colp = wpos + kx;         // pre-padded col index 0..65
        const __bf16* bp = sY + ((rowIdx * 66 + colp) * 64) + ks * 32 + kh * 8;
        v16bf bf = frag_contig(bp);
        acc[sub] = wmma_bf16(a, bf, acc[sub]);
      }
    }
  }
#pragma unroll
  for (int sub = 0; sub < 4; ++sub) {
    int ntile = nbase + 4 * sub;
    int P = ntile * 16 + n15;
    int srl = P >> 6, wpos = P & 63;
#pragma unroll
    for (int i = 0; i < 8; ++i) {
      int o = mt * 16 + kh * 8 + i;
      convout[((size_t)bl * Cc + o) * (Ss * Ww) + (size_t)(s0 + srl) * Ww + wpos] =
          acc[sub][i] + b0[o];
    }
  }
}

// ---------------------------------------------------------------------------
// Kernel 7: LayerNorm over (S,W) per (b,l,c) + gamma/beta + residual
// ---------------------------------------------------------------------------
__global__ __launch_bounds__(256) void ln_kernel(
    const float* __restrict__ x, const float* __restrict__ convout,
    const float* __restrict__ g, const float* __restrict__ bb,
    float* __restrict__ out) {
  __shared__ float ssum[256], ssq[256];
  int blc = blockIdx.x;
  int t = threadIdx.x;
  const float* cv = convout + (size_t)blc * (Ss * Ww);
  float s1 = 0.f, s2 = 0.f;
  for (int i = t; i < Ss * Ww; i += 256) {
    float v = cv[i];
    s1 += v;
    s2 += v * v;
  }
  ssum[t] = s1;
  ssq[t] = s2;
  __syncthreads();
  for (int off = 128; off > 0; off >>= 1) {
    if (t < off) {
      ssum[t] += ssum[t + off];
      ssq[t] += ssq[t + off];
    }
    __syncthreads();
  }
  float mu = ssum[0] * (1.f / (Ss * Ww));
  float var = ssq[0] * (1.f / (Ss * Ww)) - mu * mu;
  float inv = rsqrtf(var + 1e-5f);
  const float* xp = x + (size_t)blc * (Ss * Ww);
  float* op = out + (size_t)blc * (Ss * Ww);
  for (int i = t; i < Ss * Ww; i += 256) {
    op[i] = xp[i] + (cv[i] - mu) * inv * g[i] + bb[i];
  }
}

// ---------------------------------------------------------------------------
// Launcher
// ---------------------------------------------------------------------------
extern "C" void kernel_launch(void* const* d_in, const int* in_sizes, int n_in,
                              void* d_out, int out_size, void* d_ws,
                              size_t ws_size, hipStream_t stream) {
  const float* x = (const float*)d_in[0];
  const float* dt = (const float*)d_in[1];
  const float* plb = (const float*)d_in[2];
  const float* disp = (const float*)d_in[3];
  const float* w1 = (const float*)d_in[4];
  const float* b1 = (const float*)d_in[5];
  const float* w2 = (const float*)d_in[6];
  const float* b2 = (const float*)d_in[7];
  const float* fs = (const float*)d_in[8];
  const float* Ure = (const float*)d_in[9];
  const float* Uim = (const float*)d_in[10];
  const float* Vre = (const float*)d_in[11];
  const float* Vim = (const float*)d_in[12];
  const float* pWre = (const float*)d_in[13];
  const float* pWim = (const float*)d_in[14];
  const float* pbre = (const float*)d_in[15];
  const float* pbim = (const float*)d_in[16];
  const float* convr_w = (const float*)d_in[17];
  const float* convr_b = (const float*)d_in[18];
  const float* convi_w = (const float*)d_in[19];
  const float* convi_b = (const float*)d_in[20];
  const float* fuse_w = (const float*)d_in[21];
  const float* fuse_b = (const float*)d_in[22];
  const float* ln_g = (const float*)d_in[23];
  const float* ln_b = (const float*)d_in[24];
  float* out = (float*)d_out;

  const size_t BLCR = (size_t)Bb * Ll * Cc * Rr;        // 131072
  const size_t BLCSW = (size_t)Bb * Ll * Cc * Ss * Ww;  // 16777216

  char* p = (char*)d_ws;
  auto take = [&](size_t n) {
    char* q = p;
    p += (n + 255) & ~(size_t)255;
    return q;
  };
  float* lamre = (float*)take(BLCR * 4);
  float* lamim = (float*)take(BLCR * 4);
  float* zre = (float*)take(BLCR * 4);
  float* zim = (float*)take(BLCR * 4);
  float* hre = (float*)take(BLCR * 4);
  float* him = (float*)take(BLCR * 4);
  float* effr = (float*)take(32 * 32 * 9 * 4);
  float* effi = (float*)take(32 * 32 * 9 * 4);
  __bf16* wcat = (__bf16*)take(32 * 576 * 2);
  float* b0 = (float*)take(32 * 4);
  __bf16* ycat = (__bf16*)take(BLCSW * 2 * 2);  // [bl][s][w][64] Re||Im
  float* convout = (float*)take(BLCSW * 4);

  prep_kernel<<<1, 256, 0, stream>>>(pWre, pWim, convr_w, convr_b, convi_w,
                                     convi_b, fuse_w, fuse_b, effr, effi, wcat,
                                     b0);
  mlp_lam_kernel<<<Bb * Ll, 256, 0, stream>>>(x, dt, plb, disp, w1, b1, w2, b2,
                                              fs, lamre, lamim);
  encode_kernel<<<Bb * Ll * Cc, 256, 0, stream>>>(x, Ure, Uim, Vre, Vim, zre,
                                                  zim);
  scan_kernel<<<(Bb * Cc * Rr) / 256, 256, 0, stream>>>(lamre, lamim, zre, zim,
                                                        hre, him);
  decode_kernel<<<Bb * Ll * Cc, 256, 0, stream>>>(Ure, Uim, Vre, Vim, hre, him,
                                                  pbre, pbim, ycat);
  conv_kernel<<<Bb * Ll * 16, 256, 0, stream>>>(ycat, wcat, b0, convout);
  ln_kernel<<<Bb * Ll * Cc, 256, 0, stream>>>(x, convout, ln_g, ln_b, out);
}
